// GRURef_39719857553535
// MI455X (gfx1250) — matrix operand order repaired and checked by reading, not compiled, not run on hardware
//
#include <hip/hip_runtime.h>

// ---------------------------------------------------------------------------
// Types for CDNA5 WMMA
// ---------------------------------------------------------------------------
typedef __bf16 bf16_t;
typedef __attribute__((ext_vector_type(16))) __bf16 v16bf;
typedef __attribute__((ext_vector_type(8)))  __bf16 v8bf;
typedef __attribute__((ext_vector_type(8)))  float  v8f;
typedef __attribute__((ext_vector_type(4)))  float  v4f;

static __device__ __forceinline__ bf16_t f2bf(float f) { return (bf16_t)f; }
static __device__ __forceinline__ float  bf2f(bf16_t v) { return (float)v; }

static __device__ __forceinline__ v8f wmma_bf16(v16bf a, v16bf b, v8f c) {
    // D(16x16,f32) = A(16x32,bf16) x B(32x16,bf16) + C
    return __builtin_amdgcn_wmma_f32_16x16x32_bf16(
        /*neg_a=*/false, a, /*neg_b=*/false, b,
        /*c_mod=*/(short)0, c, /*reuse_a=*/false, /*reuse_b=*/false);
}

// Hardware tanh (gfx1250 TRANS op). One trailing v_nop satisfies the
// TRANS-op result-use hazard (1 independent instr before use).
static __device__ __forceinline__ float tanh_hw(float x) {
    float r;
    asm("v_tanh_f32 %0, %1\n\tv_nop" : "=v"(r) : "v"(x));
    return r;
}
static __device__ __forceinline__ float sigmoid_hw(float x) {
    return __builtin_fmaf(0.5f, tanh_hw(0.5f * x), 0.5f);
}

// Problem constants
#define BATCH 64
#define SEQT  2048
#define HID   256
#define G3    768   // 3*HID

// ---------------------------------------------------------------------------
// Kernel 0a: convert W_ih / W_hh (fp32 [768,256]) to bf16
// ---------------------------------------------------------------------------
__global__ __launch_bounds__(256) void convert_weights(
    const float* __restrict__ Wih, const float* __restrict__ Whh,
    bf16_t* __restrict__ WihB, bf16_t* __restrict__ WhhB, int n)
{
    int i = blockIdx.x * blockDim.x + threadIdx.x;
    if (i < n) {
        WihB[i] = f2bf(Wih[i]);
        WhhB[i] = f2bf(Whh[i]);
    }
}

// ---------------------------------------------------------------------------
// Kernel 0b: convert x (fp32 [BT,256]) to bf16, 8 elements per thread
// ---------------------------------------------------------------------------
__global__ __launch_bounds__(256) void convert_x(
    const float* __restrict__ x, bf16_t* __restrict__ xb)
{
    const size_t i = ((size_t)blockIdx.x * 256 + threadIdx.x) * 8;
    v4f a = *(const v4f*)(x + i);
    v4f b = *(const v4f*)(x + i + 4);
    v8bf o;
    #pragma unroll
    for (int e = 0; e < 4; ++e) { o[e] = f2bf(a[e]); o[4 + e] = f2bf(b[e]); }
    *(v8bf*)(xb + i) = o;
}

// ---------------------------------------------------------------------------
// Kernel 1: x_proj[r, g] = sum_k x[r,k]*W_ih[g,k] + b_ih[g] (+ b_hh[g] for
//   the r/z gate columns g < 512 — folded here so the scan's critical path
//   skips those adds).  Output bf16 [BT, 768].
//   Block = 256 threads (8 waves): 16 rows; wave w owns 96 columns.
// ---------------------------------------------------------------------------
__global__ __launch_bounds__(256) void xproj_gemm(
    const bf16_t* __restrict__ xb,     // [BT, 256] bf16
    const bf16_t* __restrict__ WihB,   // [768, 256] bf16 row-major
    const float*  __restrict__ b_ih,   // [768]
    const float*  __restrict__ b_hh,   // [768]
    bf16_t*       __restrict__ xp)     // [BT, 768] bf16
{
    const int rowbase = blockIdx.x * 16;
    const int wave = threadIdx.x >> 5;
    const int lane = threadIdx.x & 31;
    const int lo = lane & 15;
    const int hi = lane >> 4;
    const int colbase = wave * 96;

    v8f acc[6];
    #pragma unroll
    for (int nt = 0; nt < 6; ++nt) acc[nt] = (v8f){0,0,0,0,0,0,0,0};

    const bf16_t* xr = xb + (size_t)(rowbase + lo) * 256;

    #pragma unroll
    for (int kb = 0; kb < 8; ++kb) {
        // A fragment (16x32): lane row = lo; halves 0..7 = K k0.., 8..15 = K k0+8..
        const int k0 = kb * 32 + hi * 16;
        v8bf l0 = *(const v8bf*)(xr + k0);
        v8bf l1 = *(const v8bf*)(xr + k0 + 8);
        v16bf a;
        #pragma unroll
        for (int e = 0; e < 8; ++e) { a[e] = l0[e]; a[8 + e] = l1[e]; }

        // B fragments: B[k,n] = W[n,k]; lane col n = lo; 32 contiguous bytes.
        #pragma unroll
        for (int nt = 0; nt < 6; ++nt) {
            const bf16_t* wr = WihB + (size_t)(colbase + nt * 16 + lo) * 256
                                    + kb * 32 + hi * 16;
            v16bf b = *(const v16bf*)wr;
            acc[nt] = wmma_bf16(a, b, acc[nt]);
        }
    }

    // Epilogue: C layout: vgpr v -> M = v + 8*hi, N = lo (per 16-wide tile)
    #pragma unroll
    for (int nt = 0; nt < 6; ++nt) {
        const int col = colbase + nt * 16 + lo;
        float bias = b_ih[col];
        if (col < 512) bias += b_hh[col];   // fold r/z recurrent biases
        #pragma unroll
        for (int v = 0; v < 8; ++v) {
            const int row = rowbase + v + 8 * hi;
            xp[(size_t)row * G3 + col] = f2bf(acc[nt][v] + bias);
        }
    }
}

// ---------------------------------------------------------------------------
// Swizzled LDS layout for h: stored directly in WMMA-A-fragment order so a
// lane's whole 16-half A fragment for K-chunk kb is one contiguous 32B read.
//   element (m, c):  kb = c>>5, k = c&31, hi_t = (k>>4)&1,
//                    e = k&15
//   idx = (kb*2 + hi_t)*256 + m*16 + e         (halves; 4096 per buffer)
// ---------------------------------------------------------------------------
static __device__ __forceinline__ int hswz(int m, int c) {
    const int kb = c >> 5;
    const int k  = c & 31;
    const int hit = (k >> 4) & 1;
    const int e  = k & 15;
    return (kb * 2 + hit) * 256 + m * 16 + e;
}

// ---------------------------------------------------------------------------
// Kernel 2: sequential GRU scan. Batch rows are independent -> 4 WGs x 16
//   rows, no inter-WG sync. 512 thr = 16 waves; wave w owns h columns
//   [w*16, w*16+16) and gate columns {c, c+256, c+512}: gate fusion needs no
//   cross-wave exchange. h lives in LDS (bf16, swizzled, ping-pong) as the
//   shared WMMA A operand; old h for the z*h term stays in registers.
//   All 24 W_hh B-fragments (time-invariant) are preloaded into VGPRs, so
//   the per-step body is 16 ds_read_b128 + 24 register-operand WMMAs + gates.
//   Only 4 WGs exist -> occupancy is irrelevant, VGPR pressure is free.
// ---------------------------------------------------------------------------
__global__ __launch_bounds__(512) void gru_scan(
    const bf16_t* __restrict__ xp,     // [BT, 768] bf16 (biases folded r/z)
    const bf16_t* __restrict__ WhhB,   // [768, 256] bf16 row-major
    const float*  __restrict__ b_hh,   // [768]
    const float*  __restrict__ h0,     // [1, 64, 256] f32
    float*        __restrict__ y,      // [64, 2048, 256] f32
    float*        __restrict__ hn)     // [64, 256] f32
{
    __shared__ __align__(32) bf16_t hbuf[2][16 * 256];

    const int bbase = blockIdx.x * 16;       // 4 blocks
    const int wave  = threadIdx.x >> 5;      // 16 waves
    const int lane  = threadIdx.x & 31;
    const int lo = lane & 15;
    const int hi = lane >> 4;
    const int col = wave * 16 + lo;          // this lane's h column

    // init LDS h (bf16, swizzled) cooperatively
    for (int i = threadIdx.x; i < 16 * 256; i += 512) {
        const int m = i >> 8, c = i & 255;
        hbuf[0][hswz(m, c)] = f2bf(h0[(size_t)(bbase + m) * 256 + c]);
    }
    // register h: rows v + 8*hi, fixed col (matches C-tile layout)
    float hreg[8];
    #pragma unroll
    for (int v = 0; v < 8; ++v)
        hreg[v] = h0[(size_t)(bbase + v + 8 * hi) * 256 + col];

    const float bN = b_hh[512 + col];        // only the n-gate bias survives

    // per-lane constant swizzle base for h writes (col fixed -> kb/e fixed)
    const int kc  = col >> 5;
    const int kk  = col & 31;
    const int hit = (kk >> 4) & 1;
    const int ec  = kk & 15;
    const int wbase = (kc * 2 + hit) * 256 + ec;   // + m*16

    // ---- Preload all time-invariant W_hh B fragments into registers ----
    const bf16_t* wR = WhhB + (size_t)(col)       * 256;
    const bf16_t* wZ = WhhB + (size_t)(col + 256) * 256;
    const bf16_t* wN = WhhB + (size_t)(col + 512) * 256;
    v16bf BR[8], BZ[8], BN[8];               // 24 x 8 VGPRs = 192 VGPRs
    #pragma unroll
    for (int kb = 0; kb < 8; ++kb) {
        const int wk = kb * 32 + hi * 16;
        BR[kb] = *(const v16bf*)(wR + wk);
        BZ[kb] = *(const v16bf*)(wZ + wk);
        BN[kb] = *(const v16bf*)(wN + wk);
    }

    __syncthreads();

    for (int t = 0; t < SEQT; ++t) {
        const int cur = t & 1;

        // Prefetch next timestep's x_proj rows while the WMMAs run.
        if ((t + 1 < SEQT) && (lo == 0)) {
            #pragma unroll
            for (int v = 0; v < 8; ++v) {
                const size_t ro = ((size_t)(bbase + v + 8 * hi) * SEQT + (t + 1)) * G3;
                __builtin_prefetch(xp + ro + wave * 16, 0, 1);
                __builtin_prefetch(xp + ro + 256 + wave * 16, 0, 1);
                __builtin_prefetch(xp + ro + 512 + wave * 16, 0, 1);
            }
        }

        v8f aR = (v8f){0,0,0,0,0,0,0,0};
        v8f aZ = (v8f){0,0,0,0,0,0,0,0};
        v8f aN = (v8f){0,0,0,0,0,0,0,0};

        #pragma unroll
        for (int kb = 0; kb < 8; ++kb) {
            // one contiguous 32B A fragment thanks to the swizzled layout
            v16bf a = *(const v16bf*)&hbuf[cur][(kb * 2 + hi) * 256 + lo * 16];
            aR = wmma_bf16(a, BR[kb], aR);
            aZ = wmma_bf16(a, BZ[kb], aZ);
            aN = wmma_bf16(a, BN[kb], aN);
        }

        // Gates + state update. Row per vgpr v: m = v + 8*hi.
        const int nxt = cur ^ 1;
        #pragma unroll
        for (int v = 0; v < 8; ++v) {
            const int m = v + 8 * hi;
            const int b = bbase + m;
            const size_t ro = ((size_t)b * SEQT + t) * G3;
            const float ir  = bf2f(xp[ro + col]);          // has b_ih+b_hh
            const float iz  = bf2f(xp[ro + 256 + col]);    // has b_ih+b_hh
            const float in_ = bf2f(xp[ro + 512 + col]);    // has b_ih only

            const float r = sigmoid_hw(ir + aR[v]);
            const float z = sigmoid_hw(iz + aZ[v]);
            const float n = tanh_hw(__builtin_fmaf(r, aN[v] + bN, in_));
            const float hnew = n + z * (hreg[v] - n);
            hreg[v] = hnew;

            y[((size_t)b * SEQT + t) * HID + col] = hnew;
            hbuf[nxt][wbase + m * 16] = f2bf(hnew);
        }
        __syncthreads();
    }

    #pragma unroll
    for (int v = 0; v < 8; ++v)
        hn[(size_t)(bbase + v + 8 * hi) * 256 + col] = hreg[v];
}

// ---------------------------------------------------------------------------
// Launch
// ---------------------------------------------------------------------------
extern "C" void kernel_launch(void* const* d_in, const int* in_sizes, int n_in,
                              void* d_out, int out_size, void* d_ws, size_t ws_size,
                              hipStream_t stream) {
    const float* x    = (const float*)d_in[0];   // [64, 2048, 256]
    const float* h0   = (const float*)d_in[1];   // [1, 64, 256]
    const float* W_ih = (const float*)d_in[2];   // [768, 256]
    const float* b_ih = (const float*)d_in[3];   // [768]
    const float* W_hh = (const float*)d_in[4];   // [768, 256]
    const float* b_hh = (const float*)d_in[5];   // [768]

    float* y  = (float*)d_out;                        // [64, 2048, 256]
    float* hn = y + (size_t)BATCH * SEQT * HID;       // [1, 64, 256]

    char* ws = (char*)d_ws;
    bf16_t* WihB = (bf16_t*)(ws);                     //    393,216 B
    bf16_t* WhhB = (bf16_t*)(ws + 393216);            //    393,216 B
    bf16_t* xb   = (bf16_t*)(ws + 786432);            // 67,108,864 B
    bf16_t* xp   = (bf16_t*)(ws + 786432 + 67108864); // 201,326,592 B

    convert_weights<<<768, 256, 0, stream>>>(W_ih, W_hh, WihB, WhhB, G3 * HID);

    // 33,554,432 x elements / (256 thr * 8 per thr) = 16384 blocks
    convert_x<<<16384, 256, 0, stream>>>(x, xb);

    // BT/16 = 8192 row-tiles
    xproj_gemm<<<(BATCH * SEQT) / 16, 256, 0, stream>>>(xb, WihB, b_ih, b_hh, xp);

    gru_scan<<<BATCH / 16, 512, 0, stream>>>(xp, WhhB, b_hh, h0, y, hn);
}
